// RNNModel_8040178778157
// MI455X (gfx1250) — matrix operand order repaired and checked by reading, not compiled, not run on hardware
//
#include <hip/hip_runtime.h>
#include <hip/hip_bf16.h>

// ---- types ----
typedef __bf16 bf16_t;
typedef bf16_t v16bf __attribute__((ext_vector_type(16)));
typedef bf16_t v8bf  __attribute__((ext_vector_type(8)));
typedef float  v8f   __attribute__((ext_vector_type(8)));

#define B_   64
#define S_   512
#define H_   1024
#define NGRP 64            // hidden groups of 16 -> 64 workgroups
#define KT_  64            // 64 k-tiles of 32 over concat K = [x|h]
#define FRAG_ELEMS 512     // 32 lanes * 16 bf16 per B fragment

// ---------------- prep kernel 0: bias combine, zero h double-buffer, zero barrier
__global__ void k_init(float* __restrict__ bias, const float* __restrict__ b_ih,
                       const float* __restrict__ b_hh, unsigned* __restrict__ bar,
                       unsigned* __restrict__ hzero /* u32 view of 2x64x1024 bf16 */) {
    int i = blockIdx.x * 256 + threadIdx.x;
    if (i < 4096)  bias[i] = b_ih[i] + b_hh[i];
    if (i < 8)     bar[i] = 0u;
    if (i < 65536) hzero[i] = 0u;   // 262144 bytes = both h buffers
}

// ---------------- prep kernel 1: swizzle [W_ih | W_hh] into WMMA B-fragment order
// Layout: [group g][k-tile kt][gate][512 elems]  -> per (g,kt) one contiguous 4KB block.
// Within a fragment: element e = lane*16 + i ; n = gate*1024 + g*16 + (lane&15) ;
// k = kt*32 + ((lane>>4)<<3) + (i&7) + ((i>>3)<<4)   (lanes 0-15: K 0..7,16..23; lanes 16-31: K 8..15,24..31)
__global__ void k_wswz(bf16_t* __restrict__ wf, const float* __restrict__ Wih,
                       const float* __restrict__ Whh) {
    long idx = (long)blockIdx.x * 256 + threadIdx.x;   // < 64*64*4*512 = 8,388,608
    int e     = (int)(idx & 511);
    long f    = idx >> 9;          // ((g*64 + kt)*4 + gate)
    int gate  = (int)(f & 3);
    int kt    = (int)((f >> 2) & 63);
    int g     = (int)(f >> 8);
    int lane = e >> 4, i = e & 15;
    int n = gate * 1024 + g * 16 + (lane & 15);
    int k = kt * 32 + ((lane >> 4) << 3) + (i & 7) + ((i >> 3) << 4);
    float v = (k < 1024) ? Wih[(long)n * 1024 + k]
                         : Whh[(long)n * 1024 + (k - 1024)];
    wf[idx] = (bf16_t)v;
}

// ---------------- prep kernel 2: x f32 [B][S][I] -> bf16 [S][B][I]
__global__ void k_xcvt(bf16_t* __restrict__ xb, const float* __restrict__ x) {
    long idx = (long)blockIdx.x * 256 + threadIdx.x;   // < 33,554,432
    int i = (int)(idx & 1023);
    long r = idx >> 10;
    int b = (int)(r & 63);
    int t = (int)(r >> 6);
    xb[idx] = (bf16_t)x[(long)b * (S_ * H_) + (long)t * 1024 + i];
}

// ---- helpers ----
__device__ __forceinline__ v16bf ldfrag(const bf16_t* p0, const bf16_t* p1) {
    union { v16bf v; v8bf h[2]; } u;
    u.h[0] = *(const v8bf*)p0;
    u.h[1] = *(const v8bf*)p1;
    return u.v;
}
__device__ __forceinline__ float fsig(float x) {
    return __builtin_amdgcn_rcpf(1.0f + __expf(-x));
}
__device__ __forceinline__ float ftanh(float x) {
    float ax = fabsf(x);
    float e  = __expf(-2.0f * ax);
    float t  = (1.0f - e) * __builtin_amdgcn_rcpf(1.0f + e);
    return copysignf(t, x);
}

// ---------------- main persistent LSTM kernel: 64 WGs x 256 threads
__global__ void __launch_bounds__(256) k_lstm(
    const bf16_t* __restrict__ xb,    // [S][B][1024] bf16
    const bf16_t* __restrict__ wf,    // B fragments [grp][kt][gate][512]
    const float*  __restrict__ bias,  // 4096 (b_ih + b_hh)
    bf16_t* __restrict__ hbuf,        // 2 x [B][1024] bf16 double buffer
    unsigned* __restrict__ bar,
    float* __restrict__ hseq,         // [B][S][H] f32
    float* __restrict__ hlast,        // [B][H]
    float* __restrict__ clast)        // [B][H]
{
    const int g    = blockIdx.x;           // hidden group (16 units)
    const int w    = threadIdx.x >> 5;     // wave 0..7 -> rows w*16..w*16+15
    const int lane = threadIdx.x & 31;
    const int lo16 = lane & 15;
    const int hsel = (lane >> 4) << 3;     // 0 or 8

    float bias_g[4];
#pragma unroll
    for (int q = 0; q < 4; ++q) bias_g[q] = bias[q * 1024 + g * 16 + lo16];

    v8f c_state = {0.f, 0.f, 0.f, 0.f, 0.f, 0.f, 0.f, 0.f};

    const int M = w * 16 + lo16;           // A-fragment row
    const int j = g * 16 + lo16;           // hidden column for this lane
    // all B data for this WG: contiguous [kt][gate][512] starting here
    const bf16_t* wfg = wf + (long)g * (KT_ * 4 * FRAG_ELEMS) + lane * 16;
    const long arow = (long)M * 1024 + hsel;   // per-lane A gather offset

    for (int t = 0; t < S_; ++t) {
        v8f a0 = {0,0,0,0,0,0,0,0}, a1 = {0,0,0,0,0,0,0,0};
        v8f a2 = {0,0,0,0,0,0,0,0}, a3 = {0,0,0,0,0,0,0,0};
        const bf16_t* pax = xb + (long)t * (B_ * H_) + arow;          // x rows
        const bf16_t* pah = hbuf + (long)(t & 1) * (B_ * H_) + arow;  // h rows

        // ---- phase 1: K = x contribution (kt 0..31), loop-invariant bases ----
#pragma unroll 4
        for (int kt = 0; kt < 32; ++kt) {
            const bf16_t* pa = pax + kt * 32;
            v16bf a = ldfrag(pa, pa + 16);
            const bf16_t* pb = wfg + (long)kt * (4 * FRAG_ELEMS);
            v16bf b0 = ldfrag(pb,        pb + 8);
            v16bf b1 = ldfrag(pb +  512, pb +  520);
            v16bf b2 = ldfrag(pb + 1024, pb + 1032);
            v16bf b3 = ldfrag(pb + 1536, pb + 1544);
            a0 = __builtin_amdgcn_wmma_f32_16x16x32_bf16(false, a, false, b0, (short)0, a0, false, false);
            a1 = __builtin_amdgcn_wmma_f32_16x16x32_bf16(false, a, false, b1, (short)0, a1, false, false);
            a2 = __builtin_amdgcn_wmma_f32_16x16x32_bf16(false, a, false, b2, (short)0, a2, false, false);
            a3 = __builtin_amdgcn_wmma_f32_16x16x32_bf16(false, a, false, b3, (short)0, a3, false, false);
        }
        // ---- phase 2: K = h contribution (kt 32..63) ----
#pragma unroll 4
        for (int kt = 0; kt < 32; ++kt) {
            const bf16_t* pa = pah + kt * 32;
            v16bf a = ldfrag(pa, pa + 16);
            const bf16_t* pb = wfg + (long)(kt + 32) * (4 * FRAG_ELEMS);
            v16bf b0 = ldfrag(pb,        pb + 8);
            v16bf b1 = ldfrag(pb +  512, pb +  520);
            v16bf b2 = ldfrag(pb + 1024, pb + 1032);
            v16bf b3 = ldfrag(pb + 1536, pb + 1544);
            a0 = __builtin_amdgcn_wmma_f32_16x16x32_bf16(false, a, false, b0, (short)0, a0, false, false);
            a1 = __builtin_amdgcn_wmma_f32_16x16x32_bf16(false, a, false, b1, (short)0, a1, false, false);
            a2 = __builtin_amdgcn_wmma_f32_16x16x32_bf16(false, a, false, b2, (short)0, a2, false, false);
            a3 = __builtin_amdgcn_wmma_f32_16x16x32_bf16(false, a, false, b3, (short)0, a3, false, false);
        }

        // prefetch next timestep's x rows for this wave (predictable 2KB/row stream)
        if (t != S_ - 1) {
            const bf16_t* pnext = pax + (B_ * H_);
#pragma unroll
            for (int pf = 0; pf < 4; ++pf)
                __builtin_prefetch(pnext + pf * 256, 0, 3);
        }

        // epilogue: LSTM cell update for rows owned by this wave
        bf16_t* hw = hbuf + (long)((t + 1) & 1) * (B_ * H_);
#pragma unroll
        for (int v = 0; v < 8; ++v) {
            int row = w * 16 + v + hsel;           // C-layout: M = v (+8 for lanes 16-31)
            float iv = fsig (a0[v] + bias_g[0]);
            float fv = fsig (a1[v] + bias_g[1]);
            float gv = ftanh(a2[v] + bias_g[2]);
            float ov = fsig (a3[v] + bias_g[3]);
            float cn = fv * c_state[v] + iv * gv;
            c_state[v] = cn;
            float hv = ov * ftanh(cn);
            __builtin_nontemporal_store(hv, hseq + (long)row * (S_ * H_) + (long)t * H_ + j);
            hw[(long)row * H_ + j] = (bf16_t)hv;
            if (t == S_ - 1) {
                hlast[(long)row * H_ + j] = hv;
                clast[(long)row * H_ + j] = cn;
            }
        }

        if (t != S_ - 1) {
            __threadfence();                 // make h writes device-visible
            __syncthreads();
            if (threadIdx.x == 0) {
                __hip_atomic_fetch_add(bar, 1u, __ATOMIC_RELEASE, __HIP_MEMORY_SCOPE_AGENT);
                const unsigned target = (unsigned)(t + 1) * (unsigned)gridDim.x;
                while (__hip_atomic_load(bar, __ATOMIC_ACQUIRE, __HIP_MEMORY_SCOPE_AGENT) < target) {
                    __builtin_amdgcn_s_sleep(2);
                }
            }
            __syncthreads();
        }
    }
}

extern "C" void kernel_launch(void* const* d_in, const int* in_sizes, int n_in,
                              void* d_out, int out_size, void* d_ws, size_t ws_size,
                              hipStream_t stream) {
    const float* x    = (const float*)d_in[0];   // [64,512,1024]
    const float* W_ih = (const float*)d_in[1];   // [4096,1024]
    const float* W_hh = (const float*)d_in[2];   // [4096,1024]
    const float* b_ih = (const float*)d_in[3];   // [4096]
    const float* b_hh = (const float*)d_in[4];   // [4096]

    char* ws = (char*)d_ws;
    unsigned* bar  = (unsigned*)(ws + 0);                 // 256 B
    float*    bias = (float*)   (ws + 0x100);             // 16 KB
    bf16_t*   hbuf = (bf16_t*)  (ws + 0x8000);            // 256 KB (2x h buffers)
    bf16_t*   wf   = (bf16_t*)  (ws + 0x100000);          // 16 MB fragments
    bf16_t*   xb   = (bf16_t*)  (ws + 0x1100000);         // 64 MB bf16 x

    float* hseq  = (float*)d_out;                         // [64,512,1024]
    float* hlast = hseq + (long)B_ * S_ * H_;             // [64,1024]
    float* clast = hlast + (long)B_ * H_;                 // [64,1024]

    k_init<<<256, 256, 0, stream>>>(bias, b_ih, b_hh, bar, (unsigned*)hbuf);
    k_wswz<<<32768, 256, 0, stream>>>(wf, W_ih, W_hh);
    k_xcvt<<<131072, 256, 0, stream>>>(xb, x);
    k_lstm<<<NGRP, 256, 0, stream>>>(xb, wf, bias, hbuf, bar, hseq, hlast, clast);
}
